// EuclideanCodebook_7000796692957
// MI455X (gfx1250) — matrix operand order repaired and checked by reading, not compiled, not run on hardware
//
#include <hip/hip_runtime.h>

typedef __attribute__((ext_vector_type(16))) __bf16 v16bf;
typedef __attribute__((ext_vector_type(8)))  float  v8f;
typedef __attribute__((ext_vector_type(4)))  float  v4f;

#define DIMS   128
#define KCODES 1024
#define KC     128   // codebook codes staged in LDS per chunk
#define MB     256   // rows of x per block (8 waves x 2 M-tiles x 16 rows)

// ---------------- CDNA5 async global->LDS helpers (ASYNCcnt-tracked) ----------------
__device__ __forceinline__ void async_copy_b128(const void* gsrc, void* lds_dst) {
  // Flat pointer low 32 bits == wave-relative LDS address (ISA 10.2 aperture rules).
  unsigned lds_off = (unsigned)(unsigned long long)lds_dst;
  asm volatile("global_load_async_to_lds_b128 %0, %1, off"
               :: "v"(lds_off), "v"(gsrc) : "memory");
}
__device__ __forceinline__ void wait_async0() {
  asm volatile("s_wait_asynccnt 0" ::: "memory");
}

// ---------------- Kernel 1a: e_half[k] = 0.5*||embed[k]||^2 ----------------
__global__ __launch_bounds__(256) void vq_esq_kernel(const float* __restrict__ embed,
                                                     float* __restrict__ e_half) {
  int k = blockIdx.x * blockDim.x + threadIdx.x;
  if (k < KCODES) {
    const v4f* r = (const v4f*)(embed + (size_t)k * DIMS);
    float s = 0.f;
#pragma unroll
    for (int j = 0; j < DIMS / 4; ++j) {
      v4f v = r[j];
      s += v[0]*v[0] + v[1]*v[1] + v[2]*v[2] + v[3]*v[3];
    }
    e_half[k] = 0.5f * s;
  }
}

// ---------------- Kernel 1b: split codebook into bf16 hi/lo in workspace ----------------
__global__ __launch_bounds__(256) void vq_prep_kernel(const float* __restrict__ embed,
                                                      __bf16* __restrict__ ebf_hi,
                                                      __bf16* __restrict__ ebf_lo) {
  int i = blockIdx.x * blockDim.x + threadIdx.x;   // one v4f per thread
  if (i < KCODES * DIMS / 4) {
    v4f v = ((const v4f*)embed)[i];
#pragma unroll
    for (int e = 0; e < 4; ++e) {
      float f = v[e];
      __bf16 h = (__bf16)f;
      ebf_hi[4 * i + e] = h;
      ebf_lo[4 * i + e] = (__bf16)(f - (float)h);
    }
  }
}

// ---------------- shared device helpers ----------------
// A operand (16-bit 16x32), contiguous k order as in B: lanes 0-15 hold K=0..15, lanes 16-31 K=16..31.
__device__ __forceinline__ void load_a_frags(const float* __restrict__ x, int rowbase,
                                             int wave, int n16, int half,
                                             v16bf (&a_hi)[2][4], v16bf (&a_lo)[2][4]) {
#pragma unroll
  for (int t = 0; t < 2; ++t) {
    const int myrow = rowbase + wave * 32 + t * 16 + n16;
    const float* xr = x + (size_t)myrow * DIMS;
#pragma unroll
    for (int c = 0; c < 4; ++c) {
      const int base = 32 * c + half * 16;
      v8f f0 = *(const v8f*)(xr + base);
      v8f f1 = *(const v8f*)(xr + base + 8);
#pragma unroll
      for (int j = 0; j < 8; ++j) {
        float f = f0[j];
        __bf16 h = (__bf16)f;
        a_hi[t][c][j] = h;
        a_lo[t][c][j] = (__bf16)(f - (float)h);
        float g = f1[j];
        __bf16 h2 = (__bf16)g;
        a_hi[t][c][8 + j] = h2;
        a_lo[t][c][8 + j] = (__bf16)(g - (float)h2);
      }
    }
  }
}

// One KC-code chunk: 8 n-tiles x (4 k-chunks x 3 split products) x 2 M-tiles of WMMA.
__device__ __forceinline__ void compute_chunk(const __bf16* Bhi, const __bf16* Blo,
                                              const float* ehl, int kb, int n16, int half,
                                              const v16bf (&a_hi)[2][4], const v16bf (&a_lo)[2][4],
                                              float (&best)[2][8], int (&besti)[2][8]) {
  for (int nb = 0; nb < KC; nb += 16) {
    v8f acc0 = {0.f, 0.f, 0.f, 0.f, 0.f, 0.f, 0.f, 0.f};
    v8f acc1 = {0.f, 0.f, 0.f, 0.f, 0.f, 0.f, 0.f, 0.f};
#pragma unroll
    for (int c = 0; c < 4; ++c) {
      // B 32x16, same contiguous k order as A: lanes 0-15 K=0..15, lanes 16-31 K=16..31, N = lane%16.
      const int boff = (nb + n16) * DIMS + 32 * c + half * 16;
      v16bf bhi = *(const v16bf*)(Bhi + boff);
      v16bf blo = *(const v16bf*)(Blo + boff);
      acc0 = __builtin_amdgcn_wmma_f32_16x16x32_bf16(false, a_hi[0][c], false, bhi, (short)0, acc0, false, false);
      acc1 = __builtin_amdgcn_wmma_f32_16x16x32_bf16(false, a_hi[1][c], false, bhi, (short)0, acc1, false, false);
      acc0 = __builtin_amdgcn_wmma_f32_16x16x32_bf16(false, a_hi[0][c], false, blo, (short)0, acc0, false, false);
      acc1 = __builtin_amdgcn_wmma_f32_16x16x32_bf16(false, a_hi[1][c], false, blo, (short)0, acc1, false, false);
      acc0 = __builtin_amdgcn_wmma_f32_16x16x32_bf16(false, a_lo[0][c], false, bhi, (short)0, acc0, false, false);
      acc1 = __builtin_amdgcn_wmma_f32_16x16x32_bf16(false, a_lo[1][c], false, bhi, (short)0, acc1, false, false);
    }
    const float eh   = ehl[nb + n16];
    const int   code = kb + nb + n16;
#pragma unroll
    for (int j = 0; j < 8; ++j) {
      float s0 = 2.f * acc0[j] - eh;
      if (s0 > best[0][j]) { best[0][j] = s0; besti[0][j] = code; }
      float s1 = 2.f * acc1[j] - eh;
      if (s1 > best[1][j]) { best[1][j] = s1; besti[1][j] = code; }
    }
  }
}

__device__ __forceinline__ void finish_block(const float* __restrict__ embed,
                                             float* __restrict__ out_q, int* __restrict__ out_i,
                                             int rowbase, int tid, int wave, int n16, int half,
                                             float (&best)[2][8], int (&besti)[2][8],
                                             int* lds_idx) {
  // Reduce across the 16 lanes sharing the same rows (C layout: VGPR j -> row half*8+j).
#pragma unroll
  for (int t = 0; t < 2; ++t) {
#pragma unroll
    for (int j = 0; j < 8; ++j) {
      float v = best[t][j]; int bi = besti[t][j];
#pragma unroll
      for (int off = 1; off < 16; off <<= 1) {
        float v2 = __shfl_xor(v, off, 16);
        int   i2 = __shfl_xor(bi, off, 16);
        if (v2 > v || (v2 == v && i2 < bi)) { v = v2; bi = i2; }
      }
      best[t][j] = v; besti[t][j] = bi;
    }
  }
  if (n16 == 0) {
#pragma unroll
    for (int t = 0; t < 2; ++t)
#pragma unroll
      for (int j = 0; j < 8; ++j) {
        int row = wave * 32 + t * 16 + half * 8 + j;
        lds_idx[row] = besti[t][j];
        out_i[rowbase + row] = besti[t][j];
      }
  }
  __syncthreads();
  for (int it = tid; it < MB * DIMS / 4; it += 256) {
    int row  = it >> 5;
    int c4   = it & 31;
    int code = lds_idx[row];
    ((v4f*)(out_q + (size_t)(rowbase + row) * DIMS))[c4] =
        ((const v4f*)(embed + (size_t)code * DIMS))[c4];
  }
}

// ---------------- Kernel 2a: async double-buffered pipeline ----------------
__global__ __launch_bounds__(256) void vq_argmax_async_kernel(
    const float* __restrict__ x, const float* __restrict__ embed,
    const __bf16* __restrict__ ebf_hi, const __bf16* __restrict__ ebf_lo,
    const float* __restrict__ e_half, float* __restrict__ out_q, int* __restrict__ out_i)
{
  __shared__ __align__(32) __bf16 ldsB_hi[2][KC * DIMS];  // 2 x 32 KB
  __shared__ __align__(32) __bf16 ldsB_lo[2][KC * DIMS];  // 2 x 32 KB
  __shared__ float lds_eh[2][KC];
  __shared__ int   lds_idx[MB];

  const int tid  = threadIdx.x;
  const int wave = tid >> 5;
  const int lane = tid & 31;
  const int n16  = lane & 15;
  const int half = lane >> 4;
  const int rowbase = blockIdx.x * MB;

  v16bf a_hi[2][4], a_lo[2][4];
  load_a_frags(x, rowbase, wave, n16, half, a_hi, a_lo);

  float best[2][8];
  int   besti[2][8];
#pragma unroll
  for (int t = 0; t < 2; ++t)
#pragma unroll
    for (int j = 0; j < 8; ++j) { best[t][j] = -3.4e38f; besti[t][j] = 0; }

  // Prologue: fetch chunk 0 into buffer 0.
#pragma unroll
  for (int it = tid; it < KC * DIMS / 8; it += 256) {
    async_copy_b128(ebf_hi + it * 8, &ldsB_hi[0][it * 8]);
    async_copy_b128(ebf_lo + it * 8, &ldsB_lo[0][it * 8]);
  }
  if (tid < KC) lds_eh[0][tid] = e_half[tid];
  wait_async0();
  __syncthreads();

  for (int ci = 0; ci < KCODES / KC; ++ci) {
    const int buf = ci & 1;
    if (ci + 1 < KCODES / KC) {
      const int nxt = buf ^ 1;
      const int kb1 = (ci + 1) * KC;
      const __bf16* ghi = ebf_hi + (size_t)kb1 * DIMS;
      const __bf16* glo = ebf_lo + (size_t)kb1 * DIMS;
#pragma unroll
      for (int it = tid; it < KC * DIMS / 8; it += 256) {
        async_copy_b128(ghi + it * 8, &ldsB_hi[nxt][it * 8]);
        async_copy_b128(glo + it * 8, &ldsB_lo[nxt][it * 8]);
      }
      if (tid < KC) lds_eh[nxt][tid] = e_half[kb1 + tid];
    }
    compute_chunk(ldsB_hi[buf], ldsB_lo[buf], lds_eh[buf], ci * KC,
                  n16, half, a_hi, a_lo, best, besti);
    wait_async0();
    __syncthreads();
  }

  finish_block(embed, out_q, out_i, rowbase, tid, wave, n16, half, best, besti, lds_idx);
}

// ---------------- Kernel 2b: synchronous fallback (no workspace codebook) ----------------
__global__ __launch_bounds__(256) void vq_argmax_sync_kernel(
    const float* __restrict__ x, const float* __restrict__ embed,
    const float* __restrict__ e_half, float* __restrict__ out_q, int* __restrict__ out_i)
{
  __shared__ __align__(32) __bf16 ldsB_hi[KC * DIMS];
  __shared__ __align__(32) __bf16 ldsB_lo[KC * DIMS];
  __shared__ float lds_eh[KC];
  __shared__ int   lds_idx[MB];

  const int tid  = threadIdx.x;
  const int wave = tid >> 5;
  const int lane = tid & 31;
  const int n16  = lane & 15;
  const int half = lane >> 4;
  const int rowbase = blockIdx.x * MB;

  v16bf a_hi[2][4], a_lo[2][4];
  load_a_frags(x, rowbase, wave, n16, half, a_hi, a_lo);

  float best[2][8];
  int   besti[2][8];
#pragma unroll
  for (int t = 0; t < 2; ++t)
#pragma unroll
    for (int j = 0; j < 8; ++j) { best[t][j] = -3.4e38f; besti[t][j] = 0; }

  for (int kb = 0; kb < KCODES; kb += KC) {
    __syncthreads();
    const v4f* src = (const v4f*)(embed + (size_t)kb * DIMS);
    for (int it = tid; it < KC * DIMS / 4; it += 256) {
      v4f v = src[it];
      int o = it * 4;
#pragma unroll
      for (int e = 0; e < 4; ++e) {
        float f = v[e];
        __bf16 h = (__bf16)f;
        ldsB_hi[o + e] = h;
        ldsB_lo[o + e] = (__bf16)(f - (float)h);
      }
    }
    if (tid < KC) lds_eh[tid] = e_half[kb + tid];
    __syncthreads();
    compute_chunk(ldsB_hi, ldsB_lo, lds_eh, kb, n16, half, a_hi, a_lo, best, besti);
  }

  finish_block(embed, out_q, out_i, rowbase, tid, wave, n16, half, best, besti, lds_idx);
}

extern "C" void kernel_launch(void* const* d_in, const int* in_sizes, int n_in,
                              void* d_out, int out_size, void* d_ws, size_t ws_size,
                              hipStream_t stream) {
  const float* x     = (const float*)d_in[0];
  const float* embed = (const float*)d_in[1];
  const int N = in_sizes[0] / DIMS;                 // 16*4096 = 65536
  float* out_q  = (float*)d_out;                    // N*128 floats
  int*   out_i  = (int*)(out_q + (size_t)N * DIMS); // then N int32 indices

  const size_t off_hi   = 4096;                               // after e_half (4 KB)
  const size_t cb_bytes = (size_t)KCODES * DIMS * sizeof(__bf16);  // 256 KB each
  const size_t off_lo   = off_hi + cb_bytes;
  const size_t required = off_lo + cb_bytes;                  // ~516 KB

  float* e_half = (float*)d_ws;
  vq_esq_kernel<<<(KCODES + 255) / 256, 256, 0, stream>>>(embed, e_half);

  if (ws_size >= required) {
    __bf16* ebf_hi = (__bf16*)((char*)d_ws + off_hi);
    __bf16* ebf_lo = (__bf16*)((char*)d_ws + off_lo);
    vq_prep_kernel<<<(KCODES * DIMS / 4 + 255) / 256, 256, 0, stream>>>(embed, ebf_hi, ebf_lo);
    vq_argmax_async_kernel<<<N / MB, 256, 0, stream>>>(x, embed, ebf_hi, ebf_lo, e_half, out_q, out_i);
  } else {
    vq_argmax_sync_kernel<<<N / MB, 256, 0, stream>>>(x, embed, e_half, out_q, out_i);
  }
}
